// Refined_Attention_88845693485112
// MI455X (gfx1250) — compile-verified, matrix operand order checked by reading, not run on hardware
//
#include <hip/hip_runtime.h>
#include <hip/hip_bf16.h>

// ---------------------------------------------------------------------------
// Refined attention block for MI455X (gfx1250, wave32, WMMA).
// Heavy GEMMs: bf16 WMMA (v_wmma_f32_16x16x32_bf16), fp32 accumulate,
// 2x2 register tiling (32x32 C per wave -> 4 WMMAs per 4 fragment loads).
// Conv/BN/softmax stages are bandwidth-bound VALU kernels.
// ---------------------------------------------------------------------------

#define Bsz 32
#define Nseq 196
#define Cdim 768
#define Hh 12
#define Dh 64
#define HID 36
#define NN (196 * 196)
#define NP 224          // row/col padding for attention matrices (7 * 32)
#define RS 0.35355339059327373f  // sqrt(d^-0.5), d = 64

typedef __bf16 bf16_t;
typedef bf16_t v16bf __attribute__((ext_vector_type(16)));
typedef float  v8f   __attribute__((ext_vector_type(8)));

static __device__ __forceinline__ bf16_t f2bf(float x) {
  __hip_bfloat16 t = __float2bfloat16(x);
  bf16_t r;
  __builtin_memcpy(&r, &t, 2);
  return r;
}

// Load a 16-element bf16 A/B fragment slice for one lane:
// elements [0..7] at p, elements [8..15] at p+16 (K-halves per ISA layout).
static __device__ __forceinline__ v16bf load_frag(const bf16_t* p) {
  union { float4 f[2]; v16bf v; } u;
  u.f[0] = *reinterpret_cast<const float4*>(p);
  u.f[1] = *reinterpret_cast<const float4*>(p + 16);
  return u.v;
}

#define WMMA_BF16(a, b, c) \
  __builtin_amdgcn_wmma_f32_16x16x32_bf16(false, (a), false, (b), (short)0, (c), false, false)

// ---------------------------------------------------------------------------
// Generic batched GEMM: C[m,n] = sum_k A[m,k] * W[n,k]  (+ bias[n])
// A: [M x K] row-major (lda), W: [N x K] row-major (ldb), C: [M x N] (ldc).
// One 32x32 C supertile per wave (2x2 of 16x16), K stepped by 32.
// stM*32 / stN*32 may exceed valid extents (loads must be padded);
// stores are guarded by mValid/nValid.
// ---------------------------------------------------------------------------
__global__ void __launch_bounds__(128)
gemm_bf16_wmma(const bf16_t* __restrict__ A, const bf16_t* __restrict__ W,
               float* __restrict__ C, const float* __restrict__ bias,
               int stM, int stN, int K,
               int lda, int ldb, int ldc,
               long sA, long sB, long sC,
               int mValid, int nValid) {
  const int wave = threadIdx.x >> 5;
  const int lane = threadIdx.x & 31;
  const int tile = blockIdx.x * 4 + wave;
  if (tile >= stM * stN) return;            // wave-uniform exit
  const int tm = (tile / stN) * 32;
  const int tn = (tile % stN) * 32;

  const bf16_t* Ab = A + (long)blockIdx.y * sA;
  const bf16_t* Wb = W + (long)blockIdx.y * sB;
  float*        Cb = C + (long)blockIdx.y * sC;

  const int half = lane >> 4;               // K-half select
  const int r    = lane & 15;
  const bf16_t* pA0 = Ab + (long)(tm + r) * lda + half * 8;
  const bf16_t* pA1 = pA0 + (long)16 * lda;
  const bf16_t* pB0 = Wb + (long)(tn + r) * ldb + half * 8;
  const bf16_t* pB1 = pB0 + (long)16 * ldb;

  v8f acc00 = {}, acc01 = {}, acc10 = {}, acc11 = {};
  for (int k0 = 0; k0 < K; k0 += 32) {
    v16bf a0 = load_frag(pA0 + k0);
    v16bf a1 = load_frag(pA1 + k0);
    v16bf b0 = load_frag(pB0 + k0);
    v16bf b1 = load_frag(pB1 + k0);
    if (k0 + 32 < K) {                      // CDNA5 global_prefetch_b8
      __builtin_prefetch(pA0 + k0 + 32, 0, 1);
      __builtin_prefetch(pA1 + k0 + 32, 0, 1);
      __builtin_prefetch(pB0 + k0 + 32, 0, 1);
      __builtin_prefetch(pB1 + k0 + 32, 0, 1);
    }
    acc00 = WMMA_BF16(a0, b0, acc00);
    acc01 = WMMA_BF16(a0, b1, acc01);
    acc10 = WMMA_BF16(a1, b0, acc10);
    acc11 = WMMA_BF16(a1, b1, acc11);
  }

  const int n0 = tn + r;
  const int n1 = n0 + 16;
  const int mb = tm + half * 8;
  float bv0 = 0.0f, bv1 = 0.0f;
  if (bias != nullptr) {
    if (n0 < nValid) bv0 = bias[n0];
    if (n1 < nValid) bv1 = bias[n1];
  }
#pragma unroll
  for (int v = 0; v < 8; ++v) {
    int m0 = mb + v;
    int m1 = mb + 16 + v;
    if (m0 < mValid) {
      if (n0 < nValid) Cb[(long)m0 * ldc + n0] = acc00[v] + bv0;
      if (n1 < nValid) Cb[(long)m0 * ldc + n1] = acc01[v] + bv1;
    }
    if (m1 < mValid) {
      if (n0 < nValid) Cb[(long)m1 * ldc + n0] = acc10[v] + bv0;
      if (n1 < nValid) Cb[(long)m1 * ldc + n1] = acc11[v] + bv1;
    }
  }
}

// ---------------------------------------------------------------------------
// Elementwise / reduction helpers
// ---------------------------------------------------------------------------
__global__ void cast_bf16(const float* __restrict__ in, bf16_t* __restrict__ out, int n) {
  int i = blockIdx.x * blockDim.x + threadIdx.x;
  if (i < n) out[i] = f2bf(in[i]);
}

__global__ void zero_f32(float* p, int n) {
  int i = blockIdx.x * blockDim.x + threadIdx.x;
  if (i < n) p[i] = 0.0f;
}

// qkv [B*N, 3C] -> q/k bf16 [B,H,NP,Dh] (scaled, zero-padded rows),
//                  vT bf16 [B,H,Dh,NP] (transposed, zero-padded cols)
__global__ void pack_qkv(const float* __restrict__ qkv,
                         bf16_t* __restrict__ qb, bf16_t* __restrict__ kb,
                         bf16_t* __restrict__ vT, int total) {
  int idx = blockIdx.x * blockDim.x + threadIdx.x;
  if (idx >= total) return;                 // total = B*H*NP*Dh
  int dc = idx & 63;
  int np = (idx >> 6) % NP;
  int bh = idx / (NP * 64);
  int h = bh % Hh, b = bh / Hh;
  float qv = 0.f, kv = 0.f, vv = 0.f;
  if (np < Nseq) {
    long rowBase = (long)(b * Nseq + np) * (3 * Cdim);
    int cb = h * Dh + dc;
    qv = qkv[rowBase + cb] * RS;
    kv = qkv[rowBase + Cdim + cb] * RS;
    vv = qkv[rowBase + 2 * Cdim + cb];
  }
  long o = ((long)bh * NP + np) * Dh + dc;
  qb[o] = f2bf(qv);
  kb[o] = f2bf(kv);
  vT[((long)bh * Dh + dc) * NP + np] = f2bf(vv);
}

__global__ void softmax196(float* __restrict__ a) {
  float* p = a + (long)blockIdx.x * Nseq;
  __shared__ float red[256];
  int t = threadIdx.x;
  float v = (t < Nseq) ? p[t] : -1e30f;
  red[t] = v; __syncthreads();
  for (int s = 128; s > 0; s >>= 1) {
    if (t < s) red[t] = fmaxf(red[t], red[t + s]);
    __syncthreads();
  }
  float mx = red[0]; __syncthreads();
  float e = (t < Nseq) ? __expf(v - mx) : 0.0f;
  red[t] = e; __syncthreads();
  for (int s = 128; s > 0; s >>= 1) {
    if (t < s) red[t] += red[t + s];
    __syncthreads();
  }
  float inv = 1.0f / red[0];
  if (t < Nseq) p[t] = e * inv;
}

// 1x1 expand: y1[b,c,pix] = sum_h w[c,h] * attn[b,h,pix]
__global__ void conv_expand(const float* __restrict__ attn, const float* __restrict__ w,
                            float* __restrict__ y, int total) {
  int idx = blockIdx.x * blockDim.x + threadIdx.x;
  if (idx >= total) return;                 // total = B*NN
  int pix = idx % NN, b = idx / NN;
  float av[Hh];
#pragma unroll
  for (int h = 0; h < Hh; ++h) av[h] = attn[((long)(b * Hh + h)) * NN + pix];
  for (int c = 0; c < HID; ++c) {
    float s = 0.f;
#pragma unroll
    for (int h = 0; h < Hh; ++h) s += w[c * Hh + h] * av[h];
    y[((long)(b * HID + c)) * NN + pix] = s;
  }
}

// depthwise 3x3, SAME, zero pad
__global__ void dwconv3(const float* __restrict__ x, const float* __restrict__ w,
                        float* __restrict__ y, int total) {
  int idx = blockIdx.x * blockDim.x + threadIdx.x;
  if (idx >= total) return;                 // total = B*HID*NN
  int j = idx % Nseq;
  int i = (idx / Nseq) % Nseq;
  int bc = idx / NN;
  int c = bc % HID;
  const float* wp = w + c * 9;
  const float* xp = x + (long)bc * NN;
  float s = 0.f;
#pragma unroll
  for (int di = -1; di <= 1; ++di) {
    int ii = i + di;
    if (ii < 0 || ii >= Nseq) continue;
#pragma unroll
    for (int dj = -1; dj <= 1; ++dj) {
      int jj = j + dj;
      if (jj < 0 || jj >= Nseq) continue;
      s += wp[(di + 1) * 3 + (dj + 1)] * xp[ii * Nseq + jj];
    }
  }
  y[(long)idx] = s;
}

// 1x1 project: out[b,h,pix] = sum_c w[h,c] * y2[b,c,pix]
__global__ void conv_project(const float* __restrict__ y2, const float* __restrict__ w,
                             float* __restrict__ out, int total) {
  int idx = blockIdx.x * blockDim.x + threadIdx.x;
  if (idx >= total) return;                 // total = B*NN
  int pix = idx % NN, b = idx / NN;
  float acc[Hh];
#pragma unroll
  for (int h = 0; h < Hh; ++h) acc[h] = 0.f;
  for (int c = 0; c < HID; ++c) {
    float xv = y2[((long)(b * HID + c)) * NN + pix];
#pragma unroll
    for (int h = 0; h < Hh; ++h) acc[h] += w[h * HID + c] * xv;
  }
#pragma unroll
  for (int h = 0; h < Hh; ++h)
    out[((long)(b * Hh + h)) * NN + pix] = acc[h];
}

// training-mode BN stats: per channel c = blockIdx.y, over B * NN elements
__global__ void bn_stats(const float* __restrict__ x, float* __restrict__ stats,
                         int Cn, long perChan) {
  int c = blockIdx.y;
  __shared__ float ss[256], sq[256];
  float s = 0.f, q = 0.f;
  for (long e = (long)blockIdx.x * blockDim.x + threadIdx.x; e < perChan;
       e += (long)gridDim.x * blockDim.x) {
    long b = e / NN, pix = e % NN;
    float v = x[(b * Cn + c) * (long)NN + pix];
    s += v; q += v * v;
  }
  ss[threadIdx.x] = s; sq[threadIdx.x] = q; __syncthreads();
  for (int st = 128; st > 0; st >>= 1) {
    if (threadIdx.x < st) { ss[threadIdx.x] += ss[threadIdx.x + st];
                            sq[threadIdx.x] += sq[threadIdx.x + st]; }
    __syncthreads();
  }
  if (threadIdx.x == 0) {
    atomicAdd(&stats[2 * c],     ss[0]);
    atomicAdd(&stats[2 * c + 1], sq[0]);
  }
}

__global__ void bn_apply(float* __restrict__ x, const float* __restrict__ stats,
                         const float* __restrict__ g, const float* __restrict__ bta,
                         int Cn, int total, float invCnt, int relu6) {
  int idx = blockIdx.x * blockDim.x + threadIdx.x;
  if (idx >= total) return;
  int c = (idx / NN) % Cn;
  float mean = stats[2 * c] * invCnt;
  float var  = stats[2 * c + 1] * invCnt - mean * mean;
  float rr   = rsqrtf(var + 1e-5f);
  float v = (x[idx] - mean) * rr * g[c] + bta[c];
  if (relu6) v = fminf(fmaxf(v, 0.0f), 6.0f);
  x[idx] = v;
}

// attn_r fp32 [B,H,196,196] -> bf16 [B,H,NP,NP] zero-padded
__global__ void pack_attn_bf16(const float* __restrict__ src, bf16_t* __restrict__ dst,
                               int total) {
  int idx = blockIdx.x * blockDim.x + threadIdx.x;
  if (idx >= total) return;                 // total = B*H*NP*NP
  int np = idx % NP;
  int mp = (idx / NP) % NP;
  int bh = idx / (NP * NP);
  float v = 0.f;
  if (mp < Nseq && np < Nseq) v = src[((long)bh * Nseq + mp) * Nseq + np];
  dst[idx] = f2bf(v);
}

// out_bhnd fp32 [B,H,196,64] -> bf16 [B*196, 768] (head-interleaved)
__global__ void pack_out_bf16(const float* __restrict__ src, bf16_t* __restrict__ dst,
                              int total) {
  int idx = blockIdx.x * blockDim.x + threadIdx.x;
  if (idx >= total) return;                 // total = B*N*C
  int col = idx % Cdim, row = idx / Cdim;
  int b = row / Nseq, n = row % Nseq;
  int h = col >> 6, dc = col & 63;
  dst[idx] = f2bf(src[(((long)(b * Hh + h) * Nseq) + n) * Dh + dc]);
}

// ---------------------------------------------------------------------------
extern "C" void kernel_launch(void* const* d_in, const int* in_sizes, int n_in,
                              void* d_out, int out_size, void* d_ws, size_t ws_size,
                              hipStream_t stream) {
  (void)in_sizes; (void)n_in; (void)out_size; (void)ws_size;
  const float* x      = (const float*)d_in[0];
  const float* w_qkv  = (const float*)d_in[1];
  const float* w_exp  = (const float*)d_in[2];
  const float* bn1_g  = (const float*)d_in[3];
  const float* bn1_b  = (const float*)d_in[4];
  const float* w_dw   = (const float*)d_in[5];
  const float* bn2_g  = (const float*)d_in[6];
  const float* bn2_b  = (const float*)d_in[7];
  const float* w_pro  = (const float*)d_in[8];
  const float* bn3_g  = (const float*)d_in[9];
  const float* bn3_b  = (const float*)d_in[10];
  const float* abn_g  = (const float*)d_in[11];
  const float* abn_b  = (const float*)d_in[12];
  const float* w_proj = (const float*)d_in[13];
  const float* b_proj = (const float*)d_in[14];

  float* out    = (float*)d_out;                    // [B*N, C]
  float* attn_r = out + (long)Bsz * Nseq * Cdim;    // [B,H,196,196]

  // ---- workspace carving (256B aligned) ----
  size_t off = 0;
  char* base = (char*)d_ws;
  auto carve = [&](size_t bytes) -> char* {
    size_t a = (off + 255) & ~(size_t)255;
    off = a + bytes;
    return base + a;
  };
  const int M1 = Bsz * Nseq;                        // 6272
  bf16_t* xb     = (bf16_t*)carve((size_t)M1 * Cdim * 2);
  bf16_t* wqkvb  = (bf16_t*)carve((size_t)3 * Cdim * Cdim * 2);
  bf16_t* wprojb = (bf16_t*)carve((size_t)Cdim * Cdim * 2);
  float*  qkvf   = (float*)carve((size_t)M1 * 3 * Cdim * 4);
  bf16_t* qb     = (bf16_t*)carve((size_t)Bsz * Hh * NP * Dh * 2);
  bf16_t* kb     = (bf16_t*)carve((size_t)Bsz * Hh * NP * Dh * 2);
  bf16_t* vT     = (bf16_t*)carve((size_t)Bsz * Hh * Dh * NP * 2);
  float*  attnf  = (float*)carve((size_t)Bsz * Hh * NN * 4);
  float*  y1     = (float*)carve((size_t)Bsz * HID * NN * 4);
  float*  y2     = (float*)carve((size_t)Bsz * HID * NN * 4);
  bf16_t* arb    = (bf16_t*)carve((size_t)Bsz * Hh * NP * NP * 2);
  float*  stats  = (float*)carve(512 * 4);
  // reuse qkvf region for post-attention activations (qkv consumed by pack_qkv)
  float*  outb   = qkvf;                                        // [B,H,196,64]
  bf16_t* out2d  = (bf16_t*)(qkvf + (long)Bsz * Hh * Nseq * Dh);// [B*N, C]

  const float invCnt = 1.0f / (float)((long)Bsz * NN);
#define GRID1(n) dim3(((n) + 255) / 256), dim3(256), 0, stream

  // 1. casts
  cast_bf16<<<GRID1(M1 * Cdim)>>>(x, xb, M1 * Cdim);
  cast_bf16<<<GRID1(3 * Cdim * Cdim)>>>(w_qkv, wqkvb, 3 * Cdim * Cdim);
  cast_bf16<<<GRID1(Cdim * Cdim)>>>(w_proj, wprojb, Cdim * Cdim);

  // 2. QKV GEMM: [6272 x 2304] = x[6272x768] @ w_qkv^T
  {
    int sM = M1 / 32, sN = (3 * Cdim) / 32;       // 196 x 72
    gemm_bf16_wmma<<<dim3((sM * sN + 3) / 4, 1), 128, 0, stream>>>(
        xb, wqkvb, qkvf, nullptr, sM, sN, Cdim, Cdim, Cdim, 3 * Cdim,
        0, 0, 0, M1, 3 * Cdim);
  }

  // 3. split/scale/pad q,k,v
  pack_qkv<<<GRID1(Bsz * Hh * NP * 64)>>>(qkvf, qb, kb, vT, Bsz * Hh * NP * 64);

  // 4. scores: per (b,h): attn = (q*rs) @ (k*rs)^T  -> [196,196]
  gemm_bf16_wmma<<<dim3((7 * 7 + 3) / 4, Bsz * Hh), 128, 0, stream>>>(
      qb, kb, attnf, nullptr, 7, 7, Dh, Dh, Dh, Nseq,
      (long)NP * Dh, (long)NP * Dh, (long)NN, Nseq, Nseq);

  // 5. softmax over last dim
  softmax196<<<dim3(Bsz * Hh * Nseq), dim3(256), 0, stream>>>(attnf);

  // 6. DLA: 1x1 expand -> BN+ReLU6 -> dw3x3 -> BN+ReLU6 -> 1x1 project -> BN -> adaptBN
  conv_expand<<<GRID1(Bsz * NN)>>>(attnf, w_exp, y1, Bsz * NN);

  zero_f32<<<dim3(1), dim3(128), 0, stream>>>(stats, 128);
  bn_stats<<<dim3(120, HID), dim3(256), 0, stream>>>(y1, stats, HID, (long)Bsz * NN);
  bn_apply<<<GRID1(Bsz * HID * NN)>>>(y1, stats, bn1_g, bn1_b, HID, Bsz * HID * NN, invCnt, 1);

  dwconv3<<<GRID1(Bsz * HID * NN)>>>(y1, w_dw, y2, Bsz * HID * NN);

  zero_f32<<<dim3(1), dim3(128), 0, stream>>>(stats, 128);
  bn_stats<<<dim3(120, HID), dim3(256), 0, stream>>>(y2, stats, HID, (long)Bsz * NN);
  bn_apply<<<GRID1(Bsz * HID * NN)>>>(y2, stats, bn2_g, bn2_b, HID, Bsz * HID * NN, invCnt, 1);

  conv_project<<<GRID1(Bsz * NN)>>>(y2, w_pro, attn_r, Bsz * NN);

  zero_f32<<<dim3(1), dim3(128), 0, stream>>>(stats, 128);
  bn_stats<<<dim3(120, Hh), dim3(256), 0, stream>>>(attn_r, stats, Hh, (long)Bsz * NN);
  bn_apply<<<GRID1(Bsz * Hh * NN)>>>(attn_r, stats, bn3_g, bn3_b, Hh, Bsz * Hh * NN, invCnt, 0);

  zero_f32<<<dim3(1), dim3(128), 0, stream>>>(stats, 128);
  bn_stats<<<dim3(120, Hh), dim3(256), 0, stream>>>(attn_r, stats, Hh, (long)Bsz * NN);
  bn_apply<<<GRID1(Bsz * Hh * NN)>>>(attn_r, stats, abn_g, abn_b, Hh, Bsz * Hh * NN, invCnt, 0);

  // 7. out = attn_r @ v  (per b,h: [196x196] @ [196x64], K padded to NP)
  pack_attn_bf16<<<GRID1(Bsz * Hh * NP * NP)>>>(attn_r, arb, Bsz * Hh * NP * NP);
  gemm_bf16_wmma<<<dim3((7 * 2 + 3) / 4, Bsz * Hh), 128, 0, stream>>>(
      arb, vT, outb, nullptr, 7, 2, NP, NP, NP, Dh,
      (long)NP * NP, (long)Dh * NP, (long)Nseq * Dh, Nseq, Dh);

  // 8. final projection: out = out2d @ w_proj^T + b_proj
  pack_out_bf16<<<GRID1(M1 * Cdim)>>>(outb, out2d, M1 * Cdim);
  {
    int sM = M1 / 32, sN = Cdim / 32;             // 196 x 24
    gemm_bf16_wmma<<<dim3((sM * sN + 3) / 4, 1), 128, 0, stream>>>(
        out2d, wprojb, out, b_proj, sM, sN, Cdim, Cdim, Cdim, Cdim,
        0, 0, 0, M1, Cdim);
  }
#undef GRID1
}
